// AbstractConv3D_37323265802425
// MI455X (gfx1250) — matrix operand order repaired and checked by reading, not compiled, not run on hardware
//
#include <hip/hip_runtime.h>

typedef __attribute__((ext_vector_type(16))) __bf16 bf16x16;
typedef __attribute__((ext_vector_type(8)))  __bf16 bf16x8;
typedef __attribute__((ext_vector_type(8)))  float  f32x8;

#define NLVL 16
#define CIN  16
#define COUT 16
#define BATCH 2
#define TXD 16
#define TYD 4
#define TZD 2
#define HX (TXD + 2)           // 18
#define HY (TYD + 2)           // 6
#define HZ (TZD + 2)           // 4
#define NVOX (HX * HY * HZ)    // 432 staged voxels (with halo)
#define NTOT 1844282L          // sum of R^3 over all levels

__device__ __constant__ int c_RES[NLVL] = {16,18,20,22,24,27,30,34,38,42,47,52,58,64,72,80};

__global__ __launch_bounds__(256)
void conv3d_wmma_bf16_kernel(const float* __restrict__ input,
                             const float* __restrict__ weight,
                             const float* __restrict__ bias,
                             float* __restrict__ out)
{
    // LDS: input halo tile (bf16) + per-level weights (bf16, transposed, +1 zero tap)
    __shared__ __align__(16) __bf16 xs[NVOX * CIN];        // 432*16*2B = 13,824 B
    __shared__ __align__(16) __bf16 ws[28 * COUT * CIN];   // 7168*2B   = 14,336 B

    // ---- locate (level, batch, tile) from flat blockIdx (compile-time RES table) ----
    int bid = blockIdx.x;
    int l, base = 0;
    long off = 0;
    int R = 16, ntx = 1, nty = 4, ntz = 8, tiles = 32;
    for (l = 0; l < NLVL; ++l) {
        R   = c_RES[l];
        ntx = (R + TXD - 1) / TXD;
        nty = (R + TYD - 1) / TYD;
        ntz = (R + TZD - 1) / TZD;
        tiles = ntx * nty * ntz;
        int nb = BATCH * tiles;
        if (bid < base + nb) break;
        base += nb;
        off  += (long)R * R * R;
    }
    int rem = bid - base;
    int b   = rem / tiles;
    int t   = rem - b * tiles;
    int tz  = t / (ntx * nty);
    int r2  = t - tz * (ntx * nty);
    int ty  = r2 / ntx;
    int tx  = r2 - ty * ntx;
    int x0 = tx * TXD, y0 = ty * TYD, z0 = tz * TZD;

    int tid = threadIdx.x;

    // ---- stage input halo tile -> bf16 in LDS (zero pad outside grid) ----
    const float* inL = input + ((long)b * NTOT + off) * CIN;
    for (int i = tid; i < NVOX; i += 256) {
        int lz = i / (HY * HX);
        int rr = i - lz * (HY * HX);
        int ly = rr / HX;
        int lx = rr - ly * HX;
        int gz = z0 - 1 + lz, gy = y0 - 1 + ly, gx = x0 - 1 + lx;
        bool ok = ((unsigned)gz < (unsigned)R) & ((unsigned)gy < (unsigned)R) &
                  ((unsigned)gx < (unsigned)R);
        float v[16];
        if (ok) {
            const float4* p = (const float4*)(inL + ((long)(gz * R + gy) * R + gx) * CIN);
            float4 a0 = p[0], a1 = p[1], a2 = p[2], a3 = p[3];
            v[0]=a0.x; v[1]=a0.y; v[2]=a0.z; v[3]=a0.w;
            v[4]=a1.x; v[5]=a1.y; v[6]=a1.z; v[7]=a1.w;
            v[8]=a2.x; v[9]=a2.y; v[10]=a2.z; v[11]=a2.w;
            v[12]=a3.x; v[13]=a3.y; v[14]=a3.z; v[15]=a3.w;
        } else {
            #pragma unroll
            for (int j = 0; j < 16; ++j) v[j] = 0.0f;
        }
        bf16x8 lo, hi;
        #pragma unroll
        for (int j = 0; j < 8; ++j) { lo[j] = (__bf16)v[j]; hi[j] = (__bf16)v[j + 8]; }
        *(bf16x8*)&xs[i * CIN]     = lo;
        *(bf16x8*)&xs[i * CIN + 8] = hi;
    }

    // ---- stage per-level weights -> bf16 LDS as ws[tap][cout][cin]; tap 27 = zeros ----
    const float* wL = weight + (long)l * 27 * CIN * COUT;
    for (int i = tid; i < 28 * 256; i += 256) {
        int tp = i >> 8;
        int rr = i & 255;
        int co = rr >> 4;
        int ci = rr & 15;
        float v = (tp < 27) ? wL[(tp * CIN + ci) * COUT + co] : 0.0f;
        ws[(tp * COUT + co) * CIN + ci] = (__bf16)v;
    }
    __syncthreads();

    // ---- per-wave 16-voxel x-strip: D(16x16) accumulated over 14 K=32 WMMAs ----
    int wv   = tid >> 5;
    int lane = tid & 31;
    int L = lane & 15;        // M row (A) / N column (B,D)
    int h = lane >> 4;        // lane half selector
    int zl = wv >> 2;         // 0..1
    int yl = wv & 3;          // 0..3

    f32x8 acc = {};
    #pragma unroll
    for (int p = 0; p < 14; ++p) {
        const int ta0 = 2 * p;
        const int ta1 = (2 * p + 1 < 27) ? 2 * p + 1 : 26;  // alias finite data for pad
        const int dz0 = ta0 / 9, dy0 = (ta0 / 3) % 3, dx0 = ta0 % 3;
        const int dz1 = ta1 / 9, dy1 = (ta1 / 3) % 3, dx1 = ta1 % 3;
        int v0 = ((zl + dz0) * HY + (yl + dy0)) * HX + (L + dx0);
        int v1 = ((zl + dz1) * HY + (yl + dy1)) * HX + (L + dx1);
        // A 16x32 bf16 layout: lane (L,h): elems 0..7 = tap ta0 cin[8h..8h+7],
        //                                  elems 8..15 = tap ta1 cin[8h..8h+7]
        bf16x8 alo = *(const bf16x8*)&xs[v0 * CIN + 8 * h];
        bf16x8 ahi = *(const bf16x8*)&xs[v1 * CIN + 8 * h];
        bf16x16 A = __builtin_shufflevector(alo, ahi, 0,1,2,3,4,5,6,7,8,9,10,11,12,13,14,15);
        // B 32x16 bf16 layout: lane (L,h): elems 0..7 = tap 2p cin[8h..8h+7],
        //                                  elems 8..15 = tap 2p+1 cin[8h..8h+7]
        int tb = 2 * p + 1;   // 27 hits the zeroed slot
        const __bf16* wp = &ws[(2 * p * COUT + L) * CIN + 8 * h];
        bf16x8 blo = *(const bf16x8*)wp;
        bf16x8 bhi = *(const bf16x8*)&ws[(tb * COUT + L) * CIN + 8 * h];
        bf16x16 Bm = __builtin_shufflevector(blo, bhi, 0,1,2,3,4,5,6,7,8,9,10,11,12,13,14,15);
        acc = __builtin_amdgcn_wmma_f32_16x16x32_bf16(false, A, false, Bm,
                                                      (short)0, acc, false, false);
    }

    // ---- bias + store (D layout: VGPR r, lane(n+16h) -> M=r+8h, N=n) ----
    float bv = bias[l * COUT + L];
    int gy = y0 + yl, gz = z0 + zl;
    if (gy < R && gz < R) {
        float* outL = out + ((long)b * NTOT + off) * COUT;
        long rowBase = ((long)(gz * R + gy) * R);
        #pragma unroll
        for (int rI = 0; rI < 8; ++rI) {
            int gx = x0 + rI + 8 * h;
            if (gx < R)
                outL[(rowBase + gx) * COUT + L] = acc[rI] + bv;
        }
    }
}

extern "C" void kernel_launch(void* const* d_in, const int* in_sizes, int n_in,
                              void* d_out, int out_size, void* d_ws, size_t ws_size,
                              hipStream_t stream) {
    (void)in_sizes; (void)n_in; (void)d_ws; (void)ws_size; (void)out_size;
    const float* input  = (const float*)d_in[0];
    const float* weight = (const float*)d_in[1];
    const float* bias   = (const float*)d_in[2];
    float* out = (float*)d_out;

    static const int RESh[NLVL] = {16,18,20,22,24,27,30,34,38,42,47,52,58,64,72,80};
    int blocks = 0;
    for (int l = 0; l < NLVL; ++l) {
        int R = RESh[l];
        blocks += BATCH * ((R + TXD - 1) / TXD) * ((R + TYD - 1) / TYD) * ((R + TZD - 1) / TZD);
    }
    conv3d_wmma_bf16_kernel<<<blocks, 256, 0, stream>>>(input, weight, bias, out);
}
